// MultiHeadAttention_53927609368860
// MI455X (gfx1250) — matrix operand order repaired and checked by reading, not compiled, not run on hardware
//
#include <hip/hip_runtime.h>
#include <hip/hip_bf16.h>
#include <stdint.h>

typedef __attribute__((ext_vector_type(16))) _Float16 v16h;
typedef __attribute__((ext_vector_type(8)))  _Float16 v8h;
typedef __attribute__((ext_vector_type(8)))  float    v8f;
typedef __attribute__((ext_vector_type(4)))  unsigned int v4u;
typedef __attribute__((ext_vector_type(8)))  int v8i;
typedef __attribute__((ext_vector_type(4)))  int v4i;

#define F_IN  512
#define NHEAD 8
#define DHEAD 64
#define BATCH 16
#define SEQ   1024
#define NROWS (BATCH * SEQ)

#define TM 128
#define TN 128
#define TK 32
#define LDA 40   // halves, A staged row-major (TK + 8 pad), multiple of 8
#define LDB 40   // halves, B staged n-major

#define GLOBAL_AS __attribute__((address_space(1)))
#define LDS_AS    __attribute__((address_space(3)))

// ---- CDNA5 async / TDM feature detection --------------------------------
#if __has_builtin(__builtin_amdgcn_tensor_load_to_lds) && !defined(NO_TDM)
#define HAVE_TDM 1
#else
#define HAVE_TDM 0
#endif
#if __has_builtin(__builtin_amdgcn_global_load_async_to_lds_b128)
#define HAVE_ASYNC 1
#else
#define HAVE_ASYNC 0
#endif

__device__ __forceinline__ void wait_async0() {
#if __has_builtin(__builtin_amdgcn_s_wait_asynccnt)
  __builtin_amdgcn_s_wait_asynccnt(0);
#else
  asm volatile("s_wait_asynccnt 0x0" ::: "memory");
#endif
}
__device__ __forceinline__ void wait_tensor0() {
#if __has_builtin(__builtin_amdgcn_s_wait_tensorcnt)
  __builtin_amdgcn_s_wait_tensorcnt(0);
#else
  asm volatile("s_wait_tensorcnt 0x0" ::: "memory");
#endif
}

#if HAVE_ASYNC
__device__ __forceinline__ void async_copy16(const void* g, void* l) {
  __builtin_amdgcn_global_load_async_to_lds_b128(
      (GLOBAL_AS v4i*)(uintptr_t)g,
      (LDS_AS v4i*)(uint32_t)(uintptr_t)l, 0, 0);
}
#endif

#if HAVE_TDM
// TDM: load a 32-row x 64-half tile (row stride 512 halves) into LDS with
// per-row pad of 4 DWORDs (-> LDS row stride 72 halves == LDK).
// D# bitfields per CDNA5 ISA ch.8: group0 {count, lds_addr, global_addr, type=2},
// group1 {data_size=2B, pad_enable, pad_interval=32dw, pad_amount=4dw,
//         tensor_dim0=64, tensor_dim1=32, tile_dim0=64, tile_dim1=32,
//         tensor_dim0_stride=512}. Groups 2/3 unused for 2D tiles.
__device__ __forceinline__ void tdm_load_k_tile(const _Float16* gsrc,
                                                uint32_t lds_off) {
  uint64_t ga = (uint64_t)(uintptr_t)gsrc;
  v4u g0;
  g0[0] = 1u;                     // count=1, is_restore=0, gather off
  g0[1] = lds_off;                // lds_addr (bytes)
  g0[2] = (uint32_t)ga;           // global_addr[31:0]
  g0[3] = (uint32_t)((ga >> 32) & 0x01FFFFFFu) | 0x80000000u;  // addr[56:32] | type=2
  v8i g1;
  g1[0] = (int)((1u << 16) |      // data_size = 1 -> 2 bytes
                (1u << 20) |      // pad_enable
                (4u << 22) |      // pad_interval code 4 -> 32 DWORDs
                (3u << 25));      // pad_amount  code 3 -> 4 DWORDs
  g1[1] = (int)(64u << 16);       // tensor_dim0 = 64 (bits 79:48, low part)
  g1[2] = (int)(32u << 16);       // tensor_dim1 = 32 (bits 111:80, low part)
  g1[3] = (int)(64u << 16);       // tile_dim0 = 64 (bits 127:112)
  g1[4] = 32;                     // tile_dim1 = 32 (bits 143:128)
  g1[5] = 512;                    // tensor_dim0_stride = 512 (bits 207:160)
  g1[6] = 0;
  g1[7] = 0;
  v4i z4 = {0, 0, 0, 0};
  v8i z8 = {0, 0, 0, 0, 0, 0, 0, 0};
  __builtin_amdgcn_tensor_load_to_lds(g0, g1, z4, z4, z8, 0);
}
#endif

// ---- WMMA helpers -------------------------------------------------------
__device__ __forceinline__ v8f wmma_f16(v16h a, v16h b, v8f c) {
  return __builtin_amdgcn_wmma_f32_16x16x32_f16(
      /*neg_a=*/false, a, /*neg_b=*/false, b,
      /*c_mod=*/(short)0, c, /*reuse_a=*/false, /*reuse_b=*/false);
}

// A fragment (16 rows x 32 K) from row-major halves, row stride `ld` (mult of 8).
// lane&15 = M row; halves: K = kb8+0..7 and kb8+16..23, kb8 = 8*(lane>=16)
__device__ __forceinline__ v16h load_a_frag(const _Float16* base, int ld, int lane) {
  const _Float16* p = base + (lane & 15) * ld + ((lane >> 4) << 3);
  v8h lo = *(const v8h*)(p);
  v8h hi = *(const v8h*)(p + 16);
  v16h r;
#pragma unroll
  for (int i = 0; i < 8; ++i) { r[i] = lo[i]; r[i + 8] = hi[i]; }
  return r;
}

// B fragment (32 K x 16 N) from n-major halves (base[n*ld + k]).
// lane&15 = N col; halves: K = kb8+0..7 and kb8+16..23, kb8 = 8*(lane>=16)
__device__ __forceinline__ v16h load_b_frag(const _Float16* base, int ld, int lane) {
  const _Float16* p = base + (lane & 15) * ld + ((lane >> 4) << 3);
  v8h lo = *(const v8h*)(p);
  v8h hi = *(const v8h*)(p + 16);
  v16h r;
#pragma unroll
  for (int i = 0; i < 8; ++i) { r[i] = lo[i]; r[i + 8] = hi[i]; }
  return r;
}

// ---- Kernel 1: projection GEMM  Y(f16) = X(f32) @ W(f32) + b ------------
__global__ __launch_bounds__(256) void proj_gemm_kernel(
    const float* __restrict__ X, const float* __restrict__ W,
    const float* __restrict__ bias, _Float16* __restrict__ Y) {
  __shared__ __align__(16) _Float16 As[TM * LDA];
  __shared__ __align__(16) _Float16 Bs[TN * LDB];
  const int tid = threadIdx.x, lane = tid & 31, wv = tid >> 5;
  const int wm = wv & 1, wn = wv >> 1;
  const int m0 = blockIdx.y * TM, n0 = blockIdx.x * TN;

  v8f acc[4][2] = {};

  for (int kb = 0; kb < F_IN; kb += TK) {
    // stage A: 128x32 f32 -> f16, 4 float4 per thread
#pragma unroll
    for (int i = 0; i < 4; ++i) {
      int idx = tid * 4 + i;           // 0..1023
      int r = idx >> 3, c4 = idx & 7;  // 8 float4 per row
      float4 t = *(const float4*)(X + (size_t)(m0 + r) * F_IN + kb + c4 * 4);
      _Float16* d = As + r * LDA + c4 * 4;
      d[0] = (_Float16)t.x; d[1] = (_Float16)t.y;
      d[2] = (_Float16)t.z; d[3] = (_Float16)t.w;
    }
    // prefetch next A k-slab (speculative, -> global_prefetch_b8)
    if (kb + TK < F_IN)
      __builtin_prefetch(X + (size_t)(m0 + ((tid * 4) >> 3)) * F_IN + kb + TK, 0, 1);
    // stage B transposed: Bs[n][k]
#pragma unroll
    for (int i = 0; i < 4; ++i) {
      int idx = tid * 4 + i;
      int kr = idx >> 5, n4 = idx & 31;  // 32 float4 per W row
      float4 t = *(const float4*)(W + (size_t)(kb + kr) * F_IN + n0 + n4 * 4);
      Bs[(n4 * 4 + 0) * LDB + kr] = (_Float16)t.x;
      Bs[(n4 * 4 + 1) * LDB + kr] = (_Float16)t.y;
      Bs[(n4 * 4 + 2) * LDB + kr] = (_Float16)t.z;
      Bs[(n4 * 4 + 3) * LDB + kr] = (_Float16)t.w;
    }
    __syncthreads();

    v16h bf[2];
#pragma unroll
    for (int ni = 0; ni < 2; ++ni)
      bf[ni] = load_b_frag(Bs + (wn * 32 + ni * 16) * LDB, LDB, lane);
#pragma unroll
    for (int mi = 0; mi < 4; ++mi) {
      v16h af = load_a_frag(As + (wm * 64 + mi * 16) * LDA, LDA, lane);
#pragma unroll
      for (int ni = 0; ni < 2; ++ni)
        acc[mi][ni] = wmma_f16(af, bf[ni], acc[mi][ni]);
    }
    __syncthreads();
  }

  const int half = lane >> 4, nlane = lane & 15;
#pragma unroll
  for (int mi = 0; mi < 4; ++mi)
#pragma unroll
    for (int ni = 0; ni < 2; ++ni) {
      int nc = n0 + wn * 32 + ni * 16 + nlane;
      float bv = bias[nc];
#pragma unroll
      for (int e = 0; e < 8; ++e) {
        int mr = m0 + wm * 64 + mi * 16 + half * 8 + e;
        Y[(size_t)mr * F_IN + nc] = (_Float16)(acc[mi][ni][e] + bv);
      }
    }
}

// ---- Kernel 2: flash attention per (b, h, 128-query tile) ---------------
#define KBLK 32
#define LDK  72  // Ks[key][d], 64+8
#define LDV  40  // Vst[d][key], 32+8
#define LDP  40  // per-wave P staging 16x32 (+8)
#define SOFT_SCALE 0.125f  // 1/sqrt(64)

__global__ __launch_bounds__(256) void attn_kernel(
    const _Float16* __restrict__ qh, const _Float16* __restrict__ kh,
    const _Float16* __restrict__ vh, _Float16* __restrict__ ctx) {
  __shared__ __align__(16) _Float16 Ks[KBLK * LDK];
  __shared__ __align__(16) _Float16 Vst[DHEAD * LDV];
  __shared__ __align__(16) _Float16 Pls[8 * 16 * LDP];
  const int tid = threadIdx.x, lane = tid & 31, wv = tid >> 5;
  const int bid = blockIdx.x;
  const int qt = bid & 7;          // 8 query tiles of 128
  const int h  = (bid >> 3) & 7;
  const int b  = bid >> 6;
  const size_t qrow0 = (size_t)b * SEQ + qt * 128;
  const size_t krow0 = (size_t)b * SEQ;
  const int hoff = h * DHEAD;

  // q A-fragments for this wave's 16 rows: two 32-wide d chunks, live in regs
  v16h qf[2];
  {
    const _Float16* qp =
        qh + (qrow0 + wv * 16 + (lane & 15)) * F_IN + hoff + ((lane >> 4) << 3);
#pragma unroll
    for (int dk = 0; dk < 2; ++dk) {
      v8h lo = *(const v8h*)(qp + dk * 32);
      v8h hi = *(const v8h*)(qp + dk * 32 + 16);
#pragma unroll
      for (int i = 0; i < 8; ++i) { qf[dk][i] = lo[i]; qf[dk][i + 8] = hi[i]; }
    }
  }

  float rm[8], rl[8];
#pragma unroll
  for (int e = 0; e < 8; ++e) { rm[e] = -INFINITY; rl[e] = 0.0f; }
  v8f oacc[4] = {};
  _Float16* Pw = Pls + wv * 16 * LDP;

  for (int kb = 0; kb < SEQ; kb += KBLK) {
    const _Float16* kblk = kh + (krow0 + kb) * F_IN + hoff;
    // ---- stage K tile (32 keys x 64 d) into Ks[key][d] ----
#if HAVE_TDM
    if (wv == 0) tdm_load_k_tile(kblk, (uint32_t)(uintptr_t)&Ks[0]);
#else
    {
      int key = tid >> 3, c8 = tid & 7;  // 32 keys x 8 chunks of 8 halves
#if HAVE_ASYNC
      async_copy16(kblk + key * F_IN + c8 * 8, Ks + key * LDK + c8 * 8);
#else
      *(v8h*)(Ks + key * LDK + c8 * 8) =
          *(const v8h*)(kblk + key * F_IN + c8 * 8);
#endif
    }
#endif
    // ---- stage V tile transposed: Vst[d][key] ----
    {
      int key = tid >> 3, c8 = tid & 7;
      const _Float16* vp = vh + (krow0 + kb + key) * F_IN + hoff + c8 * 8;
      v8h t = *(const v8h*)vp;
#pragma unroll
      for (int j = 0; j < 8; ++j) Vst[(c8 * 8 + j) * LDV + key] = t[j];
    }
#if HAVE_TDM
    if (wv == 0) wait_tensor0();
#elif HAVE_ASYNC
    wait_async0();
#endif
    __syncthreads();

    // scores: 16q x 32keys, K-dim = d (64 -> 2 wmma steps per 16-key frag)
    v8f s[2] = {};
#pragma unroll
    for (int f = 0; f < 2; ++f)
#pragma unroll
      for (int dk = 0; dk < 2; ++dk) {
        v16h bf = load_b_frag(Ks + f * 16 * LDK + dk * 32, LDK, lane);
        s[f] = wmma_f16(qf[dk], bf, s[f]);
      }

    // online softmax in C layout: row = e + 8*(lane>=16), key = lane&15 + 16*f
#pragma unroll
    for (int e = 0; e < 8; ++e) {
      float v0 = s[0][e] * SOFT_SCALE;
      float v1 = s[1][e] * SOFT_SCALE;
      float t = fmaxf(v0, v1);
#pragma unroll
      for (int off = 1; off < 16; off <<= 1)
        t = fmaxf(t, __shfl_xor(t, off, 32));  // stays within 16-lane half
      float nm = fmaxf(rm[e], t);
      float alpha = __expf(rm[e] - nm);
      float p0 = __expf(v0 - nm);
      float p1 = __expf(v1 - nm);
      float ps = p0 + p1;
#pragma unroll
      for (int off = 1; off < 16; off <<= 1) ps += __shfl_xor(ps, off, 32);
      rl[e] = rl[e] * alpha + ps;
      rm[e] = nm;
#pragma unroll
      for (int f = 0; f < 4; ++f) oacc[f][e] *= alpha;
      int prow = ((lane >> 4) << 3) + e;
      Pw[prow * LDP + (lane & 15)]      = (_Float16)p0;
      Pw[prow * LDP + 16 + (lane & 15)] = (_Float16)p1;
    }

    // O += P(16x32) x V(32x64): per-wave LDS round trip relayouts C->A frag
    v16h pf = load_a_frag(Pw, LDP, lane);
#pragma unroll
    for (int f = 0; f < 4; ++f) {
      v16h bvf = load_b_frag(Vst + f * 16 * LDV, LDV, lane);
      oacc[f] = wmma_f16(pf, bvf, oacc[f]);
    }
    __syncthreads();
  }

  // normalize and store context as f16 in (B*L, 512) layout
#pragma unroll
  for (int f = 0; f < 4; ++f)
#pragma unroll
    for (int e = 0; e < 8; ++e) {
      size_t row = qrow0 + wv * 16 + ((lane >> 4) << 3) + e;
      int col = hoff + f * 16 + (lane & 15);
      ctx[row * F_IN + col] = (_Float16)(oacc[f][e] / rl[e]);
    }
}

// ---- Kernel 3: output GEMM  out = tanh([ctx | Q] @ Wo + bo) -------------
__global__ __launch_bounds__(256) void out_gemm_kernel(
    const _Float16* __restrict__ ctx, const float* __restrict__ Qin,
    const float* __restrict__ Wo, const float* __restrict__ bo,
    float* __restrict__ out) {
  __shared__ __align__(16) _Float16 As[TM * LDA];
  __shared__ __align__(16) _Float16 Bs[TN * LDB];
  const int tid = threadIdx.x, lane = tid & 31, wv = tid >> 5;
  const int wm = wv & 1, wn = wv >> 1;
  const int m0 = blockIdx.y * TM, n0 = blockIdx.x * TN;

  v8f acc[4][2] = {};

  for (int kb = 0; kb < 2 * F_IN; kb += TK) {
    if (kb < F_IN) {  // A tile from f16 context: 2x 16B per thread (async DMA)
#pragma unroll
      for (int i = 0; i < 2; ++i) {
        int idx = tid * 2 + i;           // 0..511
        int r = idx >> 2, c8 = idx & 3;  // 4 chunks of 8 halves per row
        const _Float16* g = ctx + (size_t)(m0 + r) * F_IN + kb + c8 * 8;
        _Float16* l = As + r * LDA + c8 * 8;
#if HAVE_ASYNC
        async_copy16(g, l);
#else
        *(v8h*)l = *(const v8h*)g;
#endif
      }
    } else {  // A tile from f32 Q, convert
#pragma unroll
      for (int i = 0; i < 4; ++i) {
        int idx = tid * 4 + i;
        int r = idx >> 3, c4 = idx & 7;
        float4 t = *(const float4*)(Qin + (size_t)(m0 + r) * F_IN + (kb - F_IN) + c4 * 4);
        _Float16* d = As + r * LDA + c4 * 4;
        d[0] = (_Float16)t.x; d[1] = (_Float16)t.y;
        d[2] = (_Float16)t.z; d[3] = (_Float16)t.w;
      }
    }
#pragma unroll
    for (int i = 0; i < 4; ++i) {
      int idx = tid * 4 + i;
      int kr = idx >> 5, n4 = idx & 31;
      float4 t = *(const float4*)(Wo + (size_t)(kb + kr) * F_IN + n0 + n4 * 4);
      Bs[(n4 * 4 + 0) * LDB + kr] = (_Float16)t.x;
      Bs[(n4 * 4 + 1) * LDB + kr] = (_Float16)t.y;
      Bs[(n4 * 4 + 2) * LDB + kr] = (_Float16)t.z;
      Bs[(n4 * 4 + 3) * LDB + kr] = (_Float16)t.w;
    }
#if HAVE_ASYNC
    wait_async0();
#endif
    __syncthreads();

    v16h bf[2];
#pragma unroll
    for (int ni = 0; ni < 2; ++ni)
      bf[ni] = load_b_frag(Bs + (wn * 32 + ni * 16) * LDB, LDB, lane);
#pragma unroll
    for (int mi = 0; mi < 4; ++mi) {
      v16h af = load_a_frag(As + (wm * 64 + mi * 16) * LDA, LDA, lane);
#pragma unroll
      for (int ni = 0; ni < 2; ++ni)
        acc[mi][ni] = wmma_f16(af, bf[ni], acc[mi][ni]);
    }
    __syncthreads();
  }

  const int half = lane >> 4, nlane = lane & 15;
#pragma unroll
  for (int mi = 0; mi < 4; ++mi)
#pragma unroll
    for (int ni = 0; ni < 2; ++ni) {
      int nc = n0 + wn * 32 + ni * 16 + nlane;
      float bv = bo[nc];
#pragma unroll
      for (int e = 0; e < 8; ++e) {
        int mr = m0 + wm * 64 + mi * 16 + half * 8 + e;
        out[(size_t)mr * F_IN + nc] = tanhf(acc[mi][ni][e] + bv);
      }
    }
}

// ---- host launcher ------------------------------------------------------
extern "C" void kernel_launch(void* const* d_in, const int* in_sizes, int n_in,
                              void* d_out, int out_size, void* d_ws, size_t ws_size,
                              hipStream_t stream) {
  (void)in_sizes; (void)n_in; (void)out_size; (void)ws_size;
  const float* Q  = (const float*)d_in[0];
  const float* K  = (const float*)d_in[1];
  const float* V  = (const float*)d_in[2];
  const float* Wq = (const float*)d_in[3];
  const float* bq = (const float*)d_in[4];
  const float* Wk = (const float*)d_in[5];
  const float* bk = (const float*)d_in[6];
  const float* Wv = (const float*)d_in[7];
  const float* bv = (const float*)d_in[8];
  const float* Wo = (const float*)d_in[9];
  const float* bo = (const float*)d_in[10];

  _Float16* qh = (_Float16*)d_ws;                      // 16 MB each
  _Float16* kh = qh + (size_t)NROWS * F_IN;
  _Float16* vh = kh + (size_t)NROWS * F_IN;
  _Float16* ch = vh + (size_t)NROWS * F_IN;

  dim3 blk(256);
  dim3 gproj(F_IN / TN, NROWS / TM);  // (4, 128)
  proj_gemm_kernel<<<gproj, blk, 0, stream>>>(Q, Wq, bq, qh);
  proj_gemm_kernel<<<gproj, blk, 0, stream>>>(K, Wk, bk, kh);
  proj_gemm_kernel<<<gproj, blk, 0, stream>>>(V, Wv, bv, vh);
  attn_kernel<<<dim3(BATCH * NHEAD * (SEQ / 128)), blk, 0, stream>>>(qh, kh, vh, ch);
  out_gemm_kernel<<<gproj, blk, 0, stream>>>(ch, Q, Wo, bo, (float*)d_out);
}
